// ChitEncoder_9294309229092
// MI455X (gfx1250) — matrix operand order repaired and checked by reading, not compiled, not run on hardware
//
#include <hip/hip_runtime.h>
#include <hip/hip_bf16.h>

#define TT 256
#define BB 32
#define DD 1024
#define HH 1024
#define N3 3072

typedef __bf16 bf16;
typedef bf16  bf16x4  __attribute__((ext_vector_type(4)));
typedef bf16  bf16x8  __attribute__((ext_vector_type(8)));
typedef bf16  bf16x16 __attribute__((ext_vector_type(16)));
typedef float v8f     __attribute__((ext_vector_type(8)));

union FragU { bf16x16 v; bf16x8 h[2]; };

__device__ __forceinline__ v8f wmma_bf16(bf16x16 a, bf16x16 b, v8f c) {
  return __builtin_amdgcn_wmma_f32_16x16x32_bf16(false, a, false, b, (short)0, c,
                                                 false, false);
}

// ---------------- prep: f32 [K,N] -> bf16 [N,K] transpose ----------------
__global__ __launch_bounds__(256)
void k_w_transpose(const float* __restrict__ W, bf16* __restrict__ WT,
                   int K, int N) {
  __shared__ float tile[32][33];
  int n0 = blockIdx.x * 32;
  int k0 = blockIdx.y * 32;
  int tx = threadIdx.x & 31;
  int ty = threadIdx.x >> 5;                 // 0..7
  for (int r = 0; r < 4; ++r) {
    int k = k0 + ty + r * 8;
    tile[ty + r * 8][tx] = W[(size_t)k * N + n0 + tx];
  }
  __syncthreads();
  for (int r = 0; r < 4; ++r) {
    int n = n0 + ty + r * 8;
    WT[(size_t)n * K + k0 + tx] = (bf16)tile[tx][ty + r * 8];
  }
}

// ---------------- prep: x f32 -> bf16 (same layout) ----------------
__global__ __launch_bounds__(256)
void k_x_convert(const float4* __restrict__ x, bf16* __restrict__ xb, int n4) {
  int i = blockIdx.x * blockDim.x + threadIdx.x;
  if (i >= n4) return;
  float4 v = x[i];
  bf16x4 o;
  o[0] = (bf16)v.x; o[1] = (bf16)v.y; o[2] = (bf16)v.z; o[3] = (bf16)v.w;
  *(bf16x4*)(xb + (size_t)i * 4) = o;
}

// ---------------- phase 1: xg = x @ kernel + b_in ----------------
// rows m = t*32 + b (A row = x[b][t][:]); xg stored [T*B, 3H] f32
// wave = 32 rows (2 M-tiles) x 64 cols (4 N-tiles): each B frag feeds 2 WMMAs
__global__ __launch_bounds__(256)
void k_xg_gemm(const bf16* __restrict__ xbf,      // [B,T,D] bf16
               const bf16* __restrict__ kT,       // [N3,D] bf16 (transposed)
               const float* __restrict__ bias,    // [2,3H] f32, row0=input bias
               float* __restrict__ xg) {          // [T*B, N3] f32
  int lane = threadIdx.x & 31;
  int wave = threadIdx.x >> 5;                   // 0..7
  int mw = wave & 1;
  int ng = wave >> 1;
  int m0 = blockIdx.x * 64 + mw * 32;            // 32 rows per wave
  int n0 = blockIdx.y * 256 + ng * 64;           // 64 cols per wave
  int lmod = lane & 15, lhi = lane >> 4;

  const bf16* arow[2];
  for (int mt = 0; mt < 2; ++mt) {
    int m = m0 + mt * 16 + lmod;
    int t = m >> 5, b = m & 31;
    arow[mt] = xbf + ((size_t)b * TT + t) * DD;
  }

  v8f acc[2][4];
  for (int mt = 0; mt < 2; ++mt)
    for (int j = 0; j < 4; ++j)
      for (int i = 0; i < 8; ++i) acc[mt][j][i] = 0.f;

  const bf16* brows[4];
  for (int j = 0; j < 4; ++j)
    brows[j] = kT + (size_t)(n0 + j * 16 + lmod) * DD;

  for (int kk = 0; kk < DD; kk += 32) {
    FragU a[2];
    for (int mt = 0; mt < 2; ++mt) {
      a[mt].h[0] = *(const bf16x8*)(arow[mt] + kk + lhi * 8);       // K kb..kb+7
      a[mt].h[1] = *(const bf16x8*)(arow[mt] + kk + lhi * 8 + 16);  // K kb+16..+23
    }
    for (int j = 0; j < 4; ++j) {
      const bf16* bp = brows[j] + kk + lhi * 8;                     // K kb..kb+7, kb+16..+23
      FragU fb;
      fb.h[0] = *(const bf16x8*)(bp);
      fb.h[1] = *(const bf16x8*)(bp + 16);
      acc[0][j] = wmma_bf16(a[0].v, fb.v, acc[0][j]);
      acc[1][j] = wmma_bf16(a[1].v, fb.v, acc[1][j]);
    }
  }
  for (int mt = 0; mt < 2; ++mt)
    for (int j = 0; j < 4; ++j) {
      int n = n0 + j * 16 + lmod;
      float bin = bias[n];
      int mbase = m0 + mt * 16 + lhi * 8;
      for (int i = 0; i < 8; ++i)
        xg[(size_t)(mbase + i) * N3 + n] = acc[mt][j][i] + bin;
    }
}

// ---------------- phase 2: sequential GRU recurrence ----------------
// one 1024-thread block per direction; h bf16 in LDS, h f32 in registers
__global__ __launch_bounds__(1024)
void k_gru_recur(const float* __restrict__ xg_f, const float* __restrict__ xg_b,
                 const bf16* __restrict__ rkT_f, const bf16* __restrict__ rkT_b,
                 const float* __restrict__ bias_f, const float* __restrict__ bias_b,
                 float* __restrict__ out) {       // [B,T,2H] ++ [B,2H]
  int dir = blockIdx.x;
  const float* xg  = dir ? xg_b : xg_f;
  const bf16*  rkT = dir ? rkT_b : rkT_f;
  const float* brec = (dir ? bias_b : bias_f) + N3;  // recurrent bias row
  float* statep = out + (size_t)BB * TT * 2 * HH;

  __shared__ bf16x8 hbfv[BB * HH / 8];  // 64KB: h as bf16, [32][1024]
  bf16* hbf = (bf16*)hbfv;

  int tid = threadIdx.x;
  int lane = tid & 31, wave = tid >> 5;   // wave 0..31 owns columns [32w,32w+32)
  int lmod = lane & 15, lhi = lane >> 4;
  int cw = wave * 32;

  for (int i = tid; i < BB * HH / 8; i += 1024) {
    bf16x8 z;
    for (int j = 0; j < 8; ++j) z[j] = (bf16)0.f;
    hbfv[i] = z;
  }

  v8f h[2][2], az[2][2], ar[2][2], ah[2][2];
  for (int mt = 0; mt < 2; ++mt)
    for (int nt = 0; nt < 2; ++nt)
      for (int i = 0; i < 8; ++i) h[mt][nt][i] = 0.f;

  float brz[2], brr[2], brh[2];
  for (int nt = 0; nt < 2; ++nt) {
    int n = cw + nt * 16 + lmod;
    brz[nt] = brec[n];
    brr[nt] = brec[HH + n];
    brh[nt] = brec[2 * HH + n];
  }
  __syncthreads();

  for (int s = 0; s < TT; ++s) {
    int t = dir ? (TT - 1 - s) : s;
    for (int mt = 0; mt < 2; ++mt)
      for (int nt = 0; nt < 2; ++nt)
        for (int i = 0; i < 8; ++i) {
          az[mt][nt][i] = 0.f; ar[mt][nt][i] = 0.f; ah[mt][nt][i] = 0.f;
        }

    // rg = h @ rkernel : A from LDS (ds_load_b128), B from L2-resident rkT
    for (int kk = 0; kk < HH; kk += 32) {
      FragU a[2];
      for (int mt = 0; mt < 2; ++mt) {
        int idx = (mt * 16 + lmod) * 128 + (kk >> 3) + lhi;
        a[mt].h[0] = hbfv[idx];
        a[mt].h[1] = hbfv[idx + 2];
      }
      for (int nt = 0; nt < 2; ++nt) {
        int ncol = cw + nt * 16 + lmod;
        const bf16* bz = rkT + (size_t)(ncol)          * HH + kk + lhi * 8;
        const bf16* br = rkT + (size_t)(HH + ncol)     * HH + kk + lhi * 8;
        const bf16* bh = rkT + (size_t)(2 * HH + ncol) * HH + kk + lhi * 8;
        FragU fb;
        fb.h[0] = *(const bf16x8*)bz; fb.h[1] = *(const bf16x8*)(bz + 16);
        az[0][nt] = wmma_bf16(a[0].v, fb.v, az[0][nt]);
        az[1][nt] = wmma_bf16(a[1].v, fb.v, az[1][nt]);
        fb.h[0] = *(const bf16x8*)br; fb.h[1] = *(const bf16x8*)(br + 16);
        ar[0][nt] = wmma_bf16(a[0].v, fb.v, ar[0][nt]);
        ar[1][nt] = wmma_bf16(a[1].v, fb.v, ar[1][nt]);
        fb.h[0] = *(const bf16x8*)bh; fb.h[1] = *(const bf16x8*)(bh + 16);
        ah[0][nt] = wmma_bf16(a[0].v, fb.v, ah[0][nt]);
        ah[1][nt] = wmma_bf16(a[1].v, fb.v, ah[1][nt]);
      }
    }
    __syncthreads();  // all waves done reading h before overwriting

    for (int mt = 0; mt < 2; ++mt)
      for (int nt = 0; nt < 2; ++nt) {
        int n = cw + nt * 16 + lmod;
        int mbase = mt * 16 + lhi * 8;
        for (int i = 0; i < 8; ++i) {
          int m = mbase + i;
          size_t xrow = ((size_t)t * BB + m) * N3;
          float xz = xg[xrow + n];
          float xr = xg[xrow + HH + n];
          float xh = xg[xrow + 2 * HH + n];
          float z = 1.f / (1.f + __expf(-(xz + az[mt][nt][i] + brz[nt])));
          float r = 1.f / (1.f + __expf(-(xr + ar[mt][nt][i] + brr[nt])));
          float g = xh + r * (ah[mt][nt][i] + brh[nt]);
          float e = __expf(-2.f * g);
          float th = (1.f - e) / (1.f + e);          // tanh(g)
          float hn = z * h[mt][nt][i] + (1.f - z) * th;
          h[mt][nt][i] = hn;
          hbf[m * HH + n] = (bf16)hn;
          out[((size_t)m * TT + t) * (2 * HH) + (size_t)dir * HH + n] = hn;
        }
      }
    __syncthreads();  // new h visible in LDS before next step
  }

  for (int mt = 0; mt < 2; ++mt)
    for (int nt = 0; nt < 2; ++nt) {
      int n = cw + nt * 16 + lmod;
      int mbase = mt * 16 + lhi * 8;
      for (int i = 0; i < 8; ++i)
        statep[(size_t)(mbase + i) * (2 * HH) + (size_t)dir * HH + n] =
            h[mt][nt][i];
    }
}

extern "C" void kernel_launch(void* const* d_in, const int* in_sizes, int n_in,
                              void* d_out, int out_size, void* d_ws,
                              size_t ws_size, hipStream_t stream) {
  const float* x        = (const float*)d_in[0];
  const float* kernel_f = (const float*)d_in[1];
  const float* rkern_f  = (const float*)d_in[2];
  const float* bias_f   = (const float*)d_in[3];
  const float* kernel_b = (const float*)d_in[4];
  const float* rkern_b  = (const float*)d_in[5];
  const float* bias_b   = (const float*)d_in[6];
  float* out = (float*)d_out;

  // workspace layout
  char* ws = (char*)d_ws;
  size_t off = 0;
  bf16* xbf = (bf16*)(ws + off);      off += (size_t)BB * TT * DD * 2;   // 16MB
  bf16* kT_f = (bf16*)(ws + off);     off += (size_t)N3 * DD * 2;        // 6MB
  bf16* kT_b = (bf16*)(ws + off);     off += (size_t)N3 * DD * 2;
  bf16* rkT_f = (bf16*)(ws + off);    off += (size_t)N3 * HH * 2;
  bf16* rkT_b = (bf16*)(ws + off);    off += (size_t)N3 * HH * 2;
  float* xg_f = (float*)(ws + off);   off += (size_t)TT * BB * N3 * 4;   // 96MB
  float* xg_b = (float*)(ws + off);   off += (size_t)TT * BB * N3 * 4;
  (void)ws_size; (void)in_sizes; (void)n_in; (void)out_size;

  // prep: convert x, transpose+convert weights
  int n4 = BB * TT * DD / 4;
  k_x_convert<<<(n4 + 255) / 256, 256, 0, stream>>>((const float4*)x, xbf, n4);
  dim3 tg(N3 / 32, DD / 32);
  k_w_transpose<<<tg, 256, 0, stream>>>(kernel_f, kT_f, DD, N3);
  k_w_transpose<<<tg, 256, 0, stream>>>(kernel_b, kT_b, DD, N3);
  k_w_transpose<<<tg, 256, 0, stream>>>(rkern_f, rkT_f, HH, N3);
  k_w_transpose<<<tg, 256, 0, stream>>>(rkern_b, rkT_b, HH, N3);

  // phase 1: big parallel GEMMs (M=8192, N=3072, K=1024) per direction
  dim3 gg((TT * BB) / 64, N3 / 256);
  k_xg_gemm<<<gg, 256, 0, stream>>>(xbf, kT_f, bias_f, xg_f);
  k_xg_gemm<<<gg, 256, 0, stream>>>(xbf, kT_b, bias_b, xg_b);

  // phase 2: sequential recurrence, one max-size workgroup per direction
  k_gru_recur<<<2, 1024, 0, stream>>>(xg_f, xg_b, rkT_f, rkT_b, bias_f, bias_b,
                                      out);
}
